// PTBlock_49486613184911
// MI455X (gfx1250) — matrix operand order repaired and checked by reading, not compiled, not run on hardware
//
#include <hip/hip_runtime.h>
#include <hip/hip_bf16.h>

// ---------------------------------------------------------------------------
// MI455X (gfx1250) fused Point-Transformer block.
// Mixed precision: bf16 operands, f32 accumulation via v_wmma_f32_16x16x32_bf16.
// One wave32 == one point's 16(neighbors) x 128(channels) tile.
// Weights + LDS staging stored in WMMA *operand order* (32B per lane per tile)
// so every ds_load_b128 sweeps the 64 LDS banks conflict-free.
// ---------------------------------------------------------------------------

typedef __bf16 v16bf __attribute__((ext_vector_type(16)));
typedef float  v8f   __attribute__((ext_vector_type(8)));

struct U8 { uint4 a, b; };   // 32 bytes == v16bf

// hardware RNE f32 -> bf16
static __device__ __forceinline__ unsigned short f2bfs(float f) {
    return __builtin_bit_cast(unsigned short, (__bf16)f);
}
static __device__ __forceinline__ unsigned pack2bf(float lo, float hi) {
    return (unsigned)f2bfs(lo) | ((unsigned)f2bfs(hi) << 16);
}
static __device__ __forceinline__ float bf2f(unsigned short s) {
    unsigned u = ((unsigned)s) << 16;
    return __builtin_bit_cast(float, u);
}

// A operand (16x32 bf16) fetched from ROW-MAJOR GLOBAL memory (row stride 256B):
// lane<16 : row=lane,    dw0-3 = K[kk*32+0..7],  dw4-7 = K[kk*32+16..23]
// lane>=16: row=lane-16, dw0-3 = K[kk*32+8..15], dw4-7 = K[kk*32+24..31]
static __device__ __forceinline__ v16bf load_A16g(const unsigned short* base, int lane, int kk) {
    int off = (lane & 15) * 256 + kk * 64 + ((lane >> 4) << 4);
    const char* p = (const char*)base + off;
    U8 t; t.a = *(const uint4*)p; t.b = *(const uint4*)(p + 32);
    return __builtin_bit_cast(v16bf, t);
}

// B operand from operand-order LDS: tile (nt,kk) = 1024B, lane record = 32B.
static __device__ __forceinline__ v16bf load_Bsw(const unsigned short* wt, int lane, int nt, int kk) {
    const char* p = (const char*)wt + (((nt << 2) + kk) << 10) + (lane << 5);
    U8 t; t.a = *(const uint4*)p; t.b = *(const uint4*)(p + 16);
    return __builtin_bit_cast(v16bf, t);
}

// A operand from operand-order LDS stage: tile kk = 1024B, lane record = 32B.
static __device__ __forceinline__ v16bf load_Asw(const unsigned short* st, int lane, int kk) {
    const char* p = (const char*)st + (kk << 10) + (lane << 5);
    U8 t; t.a = *(const uint4*)p; t.b = *(const uint4*)(p + 16);
    return __builtin_bit_cast(v16bf, t);
}

// ushort index of element (m,k) inside the operand-order A stage (16x128).
// A-operand mapping: r=k&31; r in [8,16)|[24,32) -> lane m+16; e = (r&7)+8*(r>=16)
static __device__ __forceinline__ int stA(int m, int k) {
    int kk = k >> 5, r = k & 31;
    int tl = m + (((r >> 3) & 1) << 4);
    int e = (r & 7) + ((r >> 4) << 3);
    return (kk << 9) + (tl << 4) + e;
}

static __device__ __forceinline__ v8f wmma_bf16(v16bf a, v16bf b, v8f c) {
    return __builtin_amdgcn_wmma_f32_16x16x32_bf16(false, a, false, b, (short)0, c, false, false);
}

// ---------------------------------------------------------------------------
// K0: convert 8 weight matrices (128x128 f32, [k][n]) into bf16 WMMA-B operand
// order: mat -> 32 tiles (nt*4+kk) -> 32 lanes -> 16 elems.
//   n = nt*16 + (L&15),  k = kk*32 + 16*(e>=8) + 8*(L>=16) + (e&7)
// ---------------------------------------------------------------------------
__global__ void __launch_bounds__(256) prep_weights(
    const float* __restrict__ w0, const float* __restrict__ w1,
    const float* __restrict__ w2, const float* __restrict__ w3,
    const float* __restrict__ w4, const float* __restrict__ w5,
    const float* __restrict__ w6, const float* __restrict__ w7,
    unsigned short* __restrict__ out)
{
    const float* s;
    switch (blockIdx.y) {
        case 0: s = w0; break; case 1: s = w1; break;
        case 2: s = w2; break; case 3: s = w3; break;
        case 4: s = w4; break; case 5: s = w5; break;
        case 6: s = w6; break; default: s = w7; break;
    }
    int f = blockIdx.x * 256 + threadIdx.x;       // 0..16383
    int e = f & 15;
    int L = (f >> 4) & 31;
    int t = f >> 9;                               // 0..31
    int kk = t & 3, nt = t >> 2;
    int n = nt * 16 + (L & 15);
    int k = kk * 32 + ((e >> 3) << 4) + ((L >> 4) << 3) + (e & 7);
    out[(size_t)blockIdx.y * 16384 + f] = f2bfs(s[k * 128 + n]);
}

// K1: x f32 -> bf16 (pairs)
__global__ void __launch_bounds__(256) cvt_bf16(const float* __restrict__ src,
                                                unsigned* __restrict__ dst, int n2)
{
    int t = blockIdx.x * 256 + threadIdx.x;
    if (t < n2) dst[t] = pack2bf(src[2 * t], src[2 * t + 1]);
}

// ---------------------------------------------------------------------------
// K2: H = A @ W + b  (bf16 in, bf16 out).  1 wave = 16 rows, weight in LDS.
// ---------------------------------------------------------------------------
__global__ void __launch_bounds__(256) gemm16_bf16out(
    const unsigned short* __restrict__ A, const unsigned short* __restrict__ Wt,
    const float* __restrict__ bias, unsigned short* __restrict__ Out)
{
    extern __shared__ char smem[];
    unsigned short* sw = (unsigned short*)smem;
    int tid = threadIdx.x, lane = tid & 31, wave = tid >> 5;
    { const uint4* s = (const uint4*)Wt; uint4* d = (uint4*)smem;
      for (int q = tid; q < 2048; q += 256) d[q] = s[q]; }
    __syncthreads();

    int row0 = (blockIdx.x * 8 + wave) * 16;
    const unsigned short* Ab = A + (size_t)row0 * 128;
    int ln15 = lane & 15, hi8 = (lane >> 4) << 3;
    #pragma unroll
    for (int nt = 0; nt < 8; nt++) {
        v8f acc = {};
        #pragma unroll
        for (int kk = 0; kk < 4; kk++)
            acc = wmma_bf16(load_A16g(Ab, lane, kk), load_Bsw(sw, lane, nt, kk), acc);
        int c = nt * 16 + ln15;
        float b = bias[c];
        #pragma unroll
        for (int r = 0; r < 8; r++) {
            int m = r + hi8;
            Out[(size_t)(row0 + m) * 128 + c] = f2bfs(acc[r] + b);
        }
    }
}

// ---------------------------------------------------------------------------
// K3: phi/psi/alpha = H @ {W_phi,W_psi,W_alpha} + b   (3 weights in LDS)
// ---------------------------------------------------------------------------
__global__ void __launch_bounds__(256) gemm_qkv(
    const unsigned short* __restrict__ H, const unsigned short* __restrict__ Wt3,
    const float* __restrict__ bphi, const float* __restrict__ bpsi, const float* __restrict__ bal,
    unsigned short* __restrict__ Ophi, unsigned short* __restrict__ Opsi,
    unsigned short* __restrict__ Oal)
{
    extern __shared__ char smem[];
    unsigned short* sw = (unsigned short*)smem;
    int tid = threadIdx.x, lane = tid & 31, wave = tid >> 5;
    { const uint4* s = (const uint4*)Wt3; uint4* d = (uint4*)smem;
      for (int q = tid; q < 2048; q += 256) { d[q] = s[q]; d[2048+q] = s[2048+q]; d[4096+q] = s[4096+q]; } }
    __syncthreads();

    int row0 = (blockIdx.x * 8 + wave) * 16;
    const unsigned short* Ab = H + (size_t)row0 * 128;
    int ln15 = lane & 15, hi8 = (lane >> 4) << 3;
    const float* bs[3] = { bphi, bpsi, bal };
    unsigned short* os[3] = { Ophi, Opsi, Oal };
    for (int o = 0; o < 3; o++) {
        const unsigned short* swo = sw + o * 16384;
        unsigned short* Out = os[o];
        #pragma unroll
        for (int nt = 0; nt < 8; nt++) {
            v8f acc = {};
            #pragma unroll
            for (int kk = 0; kk < 4; kk++)
                acc = wmma_bf16(load_A16g(Ab, lane, kk), load_Bsw(swo, lane, nt, kk), acc);
            int c = nt * 16 + ln15;
            float b = bs[o][c];
            #pragma unroll
            for (int r = 0; r < 8; r++) {
                int m = r + hi8;
                Out[(size_t)(row0 + m) * 128 + c] = f2bfs(acc[r] + b);
            }
        }
    }
}

// ---------------------------------------------------------------------------
// K4: fused neighborhood attention.  1 wave = 1 point (16 neighbors x 128 ch).
//   hidden_d = relu(rel@Wd1+b)           (VALU)
//   pos      = hidden_d@Wd2+b            (WMMA)
//   hg       = relu((phi-psi+pos)@Wg1+b) (WMMA)
//   logits   = hg@Wg2+b                  (WMMA)
//   softmax over 16 rows per channel, y = sum w*(alpha+pos)
// ---------------------------------------------------------------------------
__global__ void __launch_bounds__(256) attn_fused(
    const unsigned short* __restrict__ phiB, const unsigned short* __restrict__ psiB,
    const unsigned short* __restrict__ alB,  const float* __restrict__ coords,
    const int* __restrict__ nbr,
    const unsigned short* __restrict__ WtD2, const unsigned short* __restrict__ WtG1,
    const unsigned short* __restrict__ WtG2,
    const float* __restrict__ Wd1, const float* __restrict__ bd1,
    const float* __restrict__ bd2, const float* __restrict__ bg1,
    const float* __restrict__ bg2, unsigned short* __restrict__ yB)
{
    extern __shared__ char smem[];
    unsigned short* wd2 = (unsigned short*)smem;     // 3 x 16384 bf16 weights
    unsigned short* wg1 = wd2 + 16384;
    unsigned short* wg2 = wg1 + 16384;
    unsigned short* stageAll = wg2 + 16384;          // 8 waves x 16x128 bf16
    int tid = threadIdx.x, lane = tid & 31, wave = tid >> 5;
    { const uint4* s0 = (const uint4*)WtD2; const uint4* s1 = (const uint4*)WtG1;
      const uint4* s2 = (const uint4*)WtG2; uint4* d = (uint4*)smem;
      for (int q = tid; q < 2048; q += 256) { d[q] = s0[q]; d[2048+q] = s1[q]; d[4096+q] = s2[q]; } }
    __syncthreads();

    unsigned short* stage = stageAll + wave * 2048;
    unsigned* stage32 = (unsigned*)stage;
    int i = blockIdx.x * 8 + wave;
    int ln15 = lane & 15, hi8 = (lane >> 4) << 3;

    // neighbor index + relative coords (row m lives in lane m / m+16)
    int j0 = nbr[(size_t)i * 16 + ln15];
    float cix = coords[(size_t)i*3], ciy = coords[(size_t)i*3+1], ciz = coords[(size_t)i*3+2];
    float r0 = cix - coords[(size_t)j0*3];
    float r1 = ciy - coords[(size_t)j0*3+1];
    float r2 = ciz - coords[(size_t)j0*3+2];

    // hoisted per-half-wave neighbor row indices (8 bpermutes total)
    size_t jr[8];
    #pragma unroll
    for (int r = 0; r < 8; r++) jr[r] = (size_t)__shfl(j0, r + hi8) * 128;

    // delta layer 1 (3->128) in VALU; each lane owns channels {2*lane, 2*lane+1, +64}
    int cA = 2 * lane, cB = 2 * lane + 64;
    float wA0 = Wd1[cA],     wA0b = Wd1[cA+1];
    float wA1 = Wd1[128+cA], wA1b = Wd1[129+cA];
    float wA2 = Wd1[256+cA], wA2b = Wd1[257+cA];
    float wB0 = Wd1[cB],     wB0b = Wd1[cB+1];
    float wB1 = Wd1[128+cB], wB1b = Wd1[129+cB];
    float wB2 = Wd1[256+cB], wB2b = Wd1[257+cB];
    float bA = bd1[cA], bAb = bd1[cA+1], bB = bd1[cB], bBb = bd1[cB+1];
    #pragma unroll
    for (int m = 0; m < 16; m++) {
        float a0 = __shfl(r0, m), a1 = __shfl(r1, m), a2 = __shfl(r2, m);
        float h0 = fmaxf(0.f, bA  + a0*wA0  + a1*wA1  + a2*wA2);
        float h1 = fmaxf(0.f, bAb + a0*wA0b + a1*wA1b + a2*wA2b);
        stage32[stA(m, cA) >> 1] = pack2bf(h0, h1);
        float g0 = fmaxf(0.f, bB  + a0*wB0  + a1*wB1  + a2*wB2);
        float g1 = fmaxf(0.f, bBb + a0*wB0b + a1*wB1b + a2*wB2b);
        stage32[stA(m, cB) >> 1] = pack2bf(g0, g1);
    }

    // pos = hidden_d @ Wd2 + bd2  (kept in registers, C-layout)
    v8f pos[8];
    #pragma unroll
    for (int nt = 0; nt < 8; nt++) {
        v8f acc = {};
        #pragma unroll
        for (int kk = 0; kk < 4; kk++)
            acc = wmma_bf16(load_Asw(stage, lane, kk), load_Bsw(wd2, lane, nt, kk), acc);
        float b = bd2[nt*16 + ln15];
        #pragma unroll
        for (int r = 0; r < 8; r++) acc[r] += b;
        pos[nt] = acc;
    }

    // attn_in = phi_i - psi_j + pos  -> stage (bf16 A-operand order)
    #pragma unroll
    for (int nt = 0; nt < 8; nt++) {
        int c = nt*16 + ln15;
        float ph = bf2f(phiB[(size_t)i*128 + c]);
        #pragma unroll
        for (int r = 0; r < 8; r++) {
            int m = r + hi8;
            float ps = bf2f(psiB[jr[r] + c]);
            stage[stA(m, c)] = f2bfs(ph - ps + pos[nt][r]);
        }
    }

    // hidden_g = relu(attn_in @ Wg1 + bg1) -> stage
    v8f hg[8];
    #pragma unroll
    for (int nt = 0; nt < 8; nt++) {
        v8f acc = {};
        #pragma unroll
        for (int kk = 0; kk < 4; kk++)
            acc = wmma_bf16(load_Asw(stage, lane, kk), load_Bsw(wg1, lane, nt, kk), acc);
        hg[nt] = acc;
    }
    #pragma unroll
    for (int nt = 0; nt < 8; nt++) {
        int c = nt*16 + ln15; float b = bg1[c];
        #pragma unroll
        for (int r = 0; r < 8; r++) {
            int m = r + hi8;
            stage[stA(m, c)] = f2bfs(fmaxf(0.f, hg[nt][r] + b));
        }
    }

    // logits = hidden_g @ Wg2 + bg2 (reuse hg)
    #pragma unroll
    for (int nt = 0; nt < 8; nt++) {
        v8f acc = {};
        #pragma unroll
        for (int kk = 0; kk < 4; kk++)
            acc = wmma_bf16(load_Asw(stage, lane, kk), load_Bsw(wg2, lane, nt, kk), acc);
        float b = bg2[nt*16 + ln15];
        #pragma unroll
        for (int r = 0; r < 8; r++) acc[r] += b;
        hg[nt] = acc;
    }

    // per-channel softmax over 16 rows (half-wave shuffle) + weighted sum
    #pragma unroll
    for (int nt = 0; nt < 8; nt++) {
        int c = nt*16 + ln15;
        float mx = -3.402823466e38f;
        #pragma unroll
        for (int r = 0; r < 8; r++) mx = fmaxf(mx, hg[nt][r]);
        mx = fmaxf(mx, __shfl_xor(mx, 16));
        float s = 0.f;
        #pragma unroll
        for (int r = 0; r < 8; r++) { float e = __expf(hg[nt][r] - mx); hg[nt][r] = e; s += e; }
        s += __shfl_xor(s, 16);
        float inv = 1.f / s;
        float part = 0.f;
        #pragma unroll
        for (int r = 0; r < 8; r++) {
            float av = bf2f(alB[jr[r] + c]);
            part += hg[nt][r] * (av + pos[nt][r]);
        }
        part *= inv;
        part += __shfl_xor(part, 16);
        if (lane < 16)
            yB[(size_t)i*128 + c] = f2bfs(part);
    }
}

// ---------------------------------------------------------------------------
// K5: out = y @ W_down + b_down + x   (f32 out + residual)
// ---------------------------------------------------------------------------
__global__ void __launch_bounds__(256) gemm_down(
    const unsigned short* __restrict__ Yb, const unsigned short* __restrict__ Wt,
    const float* __restrict__ bias, const float* __restrict__ x,
    float* __restrict__ Out)
{
    extern __shared__ char smem[];
    unsigned short* sw = (unsigned short*)smem;
    int tid = threadIdx.x, lane = tid & 31, wave = tid >> 5;
    { const uint4* s = (const uint4*)Wt; uint4* d = (uint4*)smem;
      for (int q = tid; q < 2048; q += 256) d[q] = s[q]; }
    __syncthreads();

    int row0 = (blockIdx.x * 8 + wave) * 16;
    const unsigned short* Ab = Yb + (size_t)row0 * 128;
    int ln15 = lane & 15, hi8 = (lane >> 4) << 3;
    #pragma unroll
    for (int nt = 0; nt < 8; nt++) {
        v8f acc = {};
        #pragma unroll
        for (int kk = 0; kk < 4; kk++)
            acc = wmma_bf16(load_A16g(Ab, lane, kk), load_Bsw(sw, lane, nt, kk), acc);
        int c = nt * 16 + ln15;
        float b = bias[c];
        #pragma unroll
        for (int r = 0; r < 8; r++) {
            int m = r + hi8;
            size_t idx = (size_t)(row0 + m) * 128 + c;
            Out[idx] = acc[r] + b + x[idx];
        }
    }
}

// ---------------------------------------------------------------------------
extern "C" void kernel_launch(void* const* d_in, const int* in_sizes, int n_in,
                              void* d_out, int out_size, void* d_ws, size_t ws_size,
                              hipStream_t stream) {
    (void)in_sizes; (void)n_in; (void)out_size; (void)ws_size;
    const float* x       = (const float*)d_in[0];
    const float* coords  = (const float*)d_in[1];
    const int*   nbr     = (const int*)  d_in[2];
    const float* W_top   = (const float*)d_in[3];  const float* b_top   = (const float*)d_in[4];
    const float* W_down  = (const float*)d_in[5];  const float* b_down  = (const float*)d_in[6];
    const float* W_phi   = (const float*)d_in[7];  const float* b_phi   = (const float*)d_in[8];
    const float* W_psi   = (const float*)d_in[9];  const float* b_psi   = (const float*)d_in[10];
    const float* W_alpha = (const float*)d_in[11]; const float* b_alpha = (const float*)d_in[12];
    const float* W_d1    = (const float*)d_in[13]; const float* b_d1    = (const float*)d_in[14];
    const float* W_d2    = (const float*)d_in[15]; const float* b_d2    = (const float*)d_in[16];
    const float* W_g1    = (const float*)d_in[17]; const float* b_g1    = (const float*)d_in[18];
    const float* W_g2    = (const float*)d_in[19]; const float* b_g2    = (const float*)d_in[20];

    const size_t NC2 = (size_t)65536 * 128 * 2;          // bf16 matrix bytes (16 MB)
    char* ws = (char*)d_ws;
    unsigned short* wt   = (unsigned short*)ws;          // 8 x 32 KB operand-order bf16 weights
    unsigned short* xb   = (unsigned short*)(ws + 262144);
    unsigned short* hb   = (unsigned short*)(ws + 262144 + 1*NC2);
    unsigned short* phib = (unsigned short*)(ws + 262144 + 2*NC2);
    unsigned short* psib = (unsigned short*)(ws + 262144 + 3*NC2);
    unsigned short* alb  = (unsigned short*)(ws + 262144 + 4*NC2);
    unsigned short* yb   = (unsigned short*)(ws + 262144 + 5*NC2);

    // weight order: 0=top 1=phi 2=psi 3=alpha 4=d2 5=g1 6=g2 7=down
    prep_weights<<<dim3(64, 8), 256, 0, stream>>>(W_top, W_phi, W_psi, W_alpha,
                                                  W_d2, W_g1, W_g2, W_down, wt);
    cvt_bf16<<<16384, 256, 0, stream>>>(x, (unsigned*)xb, 65536 * 128 / 2);
    gemm16_bf16out<<<512, 256, 32768, stream>>>(xb, wt, b_top, hb);
    gemm_qkv<<<512, 256, 98304, stream>>>(hb, wt + 16384, b_phi, b_psi, b_alpha,
                                          phib, psib, alb);
    attn_fused<<<8192, 256, 131072, stream>>>(phib, psib, alb, coords, nbr,
                                              wt + 4*16384, wt + 5*16384, wt + 6*16384,
                                              W_d1, b_d1, b_d2, b_g1, b_g2, yb);
    gemm_down<<<512, 256, 32768, stream>>>(yb, wt + 7*16384, b_down, x, (float*)d_out);
}
